// MyRnn_55147380080658
// MI455X (gfx1250) — matrix operand order repaired and checked by reading, not compiled, not run on hardware
//
#include <hip/hip_runtime.h>
#include <math.h>

// ---------------------------------------------------------------------------
// Types for CDNA5 WMMA
// ---------------------------------------------------------------------------
typedef __attribute__((ext_vector_type(16))) __bf16 v16bf;
typedef __attribute__((ext_vector_type(8)))  float  v8f;

#define UNITS   2048
#define BATCH   256
#define SEQ     80
#define EMBED   100
#define EPAD    128          // EMBED padded to multiple of 32
#define KCH     (UNITS / 32) // 64 K-chunks in one 2048-wide hidden state

__device__ __forceinline__ unsigned short f2bf_bits(float f) {
    unsigned int u = __float_as_uint(f);
    unsigned int r = (u + 0x7fffu + ((u >> 16) & 1u)) >> 16;   // RNE
    return (unsigned short)r;
}

// branch-free tanh: 1 - 2/(exp(2x)+1); exact limits at +/-inf, no exec-mask ops
__device__ __forceinline__ float fast_tanh(float x) {
    return 1.0f - 2.0f / (__expf(2.0f * x) + 1.0f);
}

// ---------------------------------------------------------------------------
// A-fragment layout (16x32 bf16, 512 elem, lane-major 16 bf16 per lane):
//   lane L holds row m = L%16;  element e <-> k = (e/8)*16 + (L/16)*8 + (e%8)
// Fragments of a [256][K] activation stored [mTile][kc], 1KB each.
//
// B-fragment layout (32x16 bf16): lane L: element e <-> B[(e/8)*16 + (L/16)*8 + (e%8)][L%16]
// Fragments of a [K][2048] weight stored [ntile][kc], 1KB each.
// ---------------------------------------------------------------------------

// ---------------------------------------------------------------------------
// Embed + pack: produce bf16 A-fragments of x_t for ALL timesteps.
// Xp frag index = t*64 + mTile*4 + kc   (4 chunks: EPAD=128)
// ---------------------------------------------------------------------------
__global__ void embed_pack_kernel(const int* __restrict__ inputs,
                                  const float* __restrict__ emb,
                                  unsigned short* __restrict__ Xp) {
    int idx = blockIdx.x * blockDim.x + threadIdx.x;
    if (idx >= SEQ * 16 * 4 * 512) return;
    int r  = idx & 511;
    int f  = idx >> 9;
    int kc = f & 3;
    int mT = (f >> 2) & 15;
    int t  = f >> 6;
    int lane = r >> 4;
    int e    = r & 15;
    int m    = lane & 15;
    int kloc = ((e >> 3) << 4) + ((lane >> 4) << 3) + (e & 7);
    int k    = kc * 32 + kloc;
    int brow = mT * 16 + m;
    float v = 0.0f;
    if (k < EMBED) {
        int tok = inputs[brow * SEQ + t];
        v = emb[(size_t)tok * EMBED + k];
    }
    Xp[idx] = f2bf_bits(v);
}

// ---------------------------------------------------------------------------
// Zero the packed "current" hidden states (graph-safe: reruns every call)
// ---------------------------------------------------------------------------
__global__ void zero_hp_kernel(unsigned int* __restrict__ p0,
                               unsigned int* __restrict__ p1) {
    int idx = blockIdx.x * blockDim.x + threadIdx.x;
    if (idx >= 16 * KCH * 256) return;   // 1MB of bf16 = 256K dwords
    p0[idx] = 0u;
    p1[idx] = 0u;
}

// ---------------------------------------------------------------------------
// Pack fp32 [Ksrc x 2048] weights into bf16 B-fragments [ntile][kc-in-concat]
// ---------------------------------------------------------------------------
__global__ void pack_b_kernel(const float* __restrict__ src, int Ksrc,
                              int chunks, int kcStart, int KCtot,
                              unsigned short* __restrict__ dst) {
    int idx = blockIdx.x * blockDim.x + threadIdx.x;
    int total = (UNITS / 16) * chunks * 512;
    if (idx >= total) return;
    int r    = idx & 511;
    int frag = idx >> 9;
    int kcL  = frag % chunks;
    int nt   = frag / chunks;
    int lane = r >> 4;
    int e    = r & 15;
    int k = kcL * 32 + ((e >> 3) << 4) + ((lane >> 4) << 3) + (e & 7);
    int n = nt * 16 + (lane & 15);
    float v = (k < Ksrc) ? src[(size_t)k * UNITS + n] : 0.0f;
    dst[((size_t)nt * KCtot + kcStart + kcL) * 512 + r] = f2bf_bits(v);
}

// ---------------------------------------------------------------------------
// Fused GEMM + bias + tanh, operands fully prepacked bf16, barrier-free.
//   Out = tanh( [A0p | A1p] @ Bpack + bias ),  K = KC*32, A0 covers C0 chunks.
// KC / C0 are template constants -> all inter-tile B strides become immediate
// IOFFSETs (j*KC*1024 < 8MB) and the loop advance is one 2KB pointer bump.
// Grid 16 x 512thr (16 waves). Block owns a 128-col N-strip; wave w owns
// M rows [16w,16w+16) -> 8 WMMAs per 32-K chunk.
// 2-way unrolled with two named A registers: prefetch one chunk ahead with
// VMEM loads into fixed regs (no VALU rotation -> no WMMA hazard NOPs).
// ---------------------------------------------------------------------------
template <int KC, int C0>
__global__ __launch_bounds__(512)
void rnn_gemm_tanh(const unsigned short* __restrict__ A0p,
                   const unsigned short* __restrict__ A1p,
                   const unsigned short* __restrict__ Bpack,
                   const float* __restrict__ bias,
                   float* __restrict__ OutF,
                   unsigned short* __restrict__ OutPack) {
    static_assert((KC & 1) == 0, "KC must be even");
    __shared__ __align__(16) float lds[16 * 512];   // 2KB slab per wave

    const int tid  = threadIdx.x;
    const int lane = tid & 31;
    const int wave = tid >> 5;        // = mTile (0..15)
    const int ntB  = blockIdx.x * 8;  // 8 consecutive 16-col N tiles

    v8f acc[8];
    #pragma unroll
    for (int j = 0; j < 8; ++j) acc[j] = v8f{};

    const v16bf* a0 = (const v16bf*)A0p + ((size_t)wave * C0) * 32 + lane;
    const v16bf* a1 = (const v16bf*)A1p + ((size_t)wave * (KC - C0)) * 32 + lane;
    // block's B stream base; tile j lives at constant offset j*KC*32 v16bf
    const v16bf* bb = (const v16bf*)Bpack + ((size_t)ntB * KC) * 32 + lane;

    auto aload = [&](int kc) -> v16bf {
        return (kc < C0) ? a0[(size_t)kc * 32] : a1[(size_t)(kc - C0) * 32];
    };

    v16bf aEven = aload(0);
    for (int kc = 0; kc < KC; kc += 2) {
        v16bf aOdd = aload(kc + 1);
        #pragma unroll
        for (int j = 0; j < 8; ++j) {
            v16bf b = bb[(size_t)j * KC * 32 + (size_t)kc * 32];
            acc[j] = __builtin_amdgcn_wmma_f32_16x16x32_bf16(
                         false, aEven, false, b, (short)0, acc[j], false, false);
        }
        if (kc + 2 < KC) aEven = aload(kc + 2);
        #pragma unroll
        for (int j = 0; j < 8; ++j) {
            v16bf b = bb[(size_t)j * KC * 32 + (size_t)(kc + 1) * 32];
            acc[j] = __builtin_amdgcn_wmma_f32_16x16x32_bf16(
                         false, aOdd, false, b, (short)0, acc[j], false, false);
        }
    }

    // ---- epilogue: bias + tanh; fp32 store + packed-A store for next GEMM ----
    float* slab = &lds[wave * 512];          // private 16x32 fp32 tile
    const int nL     = lane & 15;
    const int mHalf  = (lane >> 4) << 3;     // 0 or 8 (D layout row base)
    const int mRead  = lane & 15;            // A-frag row for readback
    const int kb     = (lane >> 4) << 3;     // A-frag k sliver base
    const int mBase  = wave * 16;

    #pragma unroll
    for (int p = 0; p < 4; ++p) {            // pairs of tiles -> one 32-k chunk
        const int j0 = 2 * p, j1 = 2 * p + 1;
        const int n0 = (ntB + j0) * 16 + nL;
        const int n1 = (ntB + j1) * 16 + nL;
        const float bb0 = bias[n0];
        const float bb1 = bias[n1];
        #pragma unroll
        for (int r = 0; r < 8; ++r) {
            const float x0 = fast_tanh(acc[j0][r] + bb0);
            const float x1 = fast_tanh(acc[j1][r] + bb1);
            OutF[(size_t)(mBase + mHalf + r) * UNITS + n0] = x0;
            OutF[(size_t)(mBase + mHalf + r) * UNITS + n1] = x1;
            slab[(mHalf + r) * 32 + nL]      = x0;
            slab[(mHalf + r) * 32 + 16 + nL] = x1;
        }
        // same-wave DS ops are in-order: transpose readback needs no barrier
        const float* rp = slab + mRead * 32;
        float4 g0 = *(const float4*)(rp + kb);
        float4 g1 = *(const float4*)(rp + kb + 4);
        float4 g2 = *(const float4*)(rp + 16 + kb);
        float4 g3 = *(const float4*)(rp + 20 + kb);
        v16bf ao;
        ao[0]  = (__bf16)g0.x; ao[1]  = (__bf16)g0.y; ao[2]  = (__bf16)g0.z; ao[3]  = (__bf16)g0.w;
        ao[4]  = (__bf16)g1.x; ao[5]  = (__bf16)g1.y; ao[6]  = (__bf16)g1.z; ao[7]  = (__bf16)g1.w;
        ao[8]  = (__bf16)g2.x; ao[9]  = (__bf16)g2.y; ao[10] = (__bf16)g2.z; ao[11] = (__bf16)g2.w;
        ao[12] = (__bf16)g3.x; ao[13] = (__bf16)g3.y; ao[14] = (__bf16)g3.z; ao[15] = (__bf16)g3.w;
        const size_t frag = (size_t)wave * KCH + (ntB >> 1) + p;
        *(v16bf*)(OutPack + frag * 512 + lane * 16) = ao;
    }
}

// ---------------------------------------------------------------------------
// Final readout: out[b] = sigmoid( dot(H1[b], Wo) + bo )
// ---------------------------------------------------------------------------
__global__ void final_kernel(const float* __restrict__ H1,
                             const float* __restrict__ Wo,
                             const float* __restrict__ bo,
                             float* __restrict__ out) {
    int row = threadIdx.x;   // 0..255
    const float* h = H1 + (size_t)row * UNITS;
    float s = 0.0f;
    for (int k = 0; k < UNITS; k += 4) {
        float4 hv = *(const float4*)(h + k);
        float4 wv = *(const float4*)(Wo + k);
        s += hv.x * wv.x + hv.y * wv.y + hv.z * wv.z + hv.w * wv.w;
    }
    s += bo[0];
    out[row] = 1.0f / (1.0f + __expf(-s));
}

// ---------------------------------------------------------------------------
// Host launcher
// ---------------------------------------------------------------------------
extern "C" void kernel_launch(void* const* d_in, const int* in_sizes, int n_in,
                              void* d_out, int out_size, void* d_ws, size_t ws_size,
                              hipStream_t stream) {
    const int*   inputs = (const int*)  d_in[0];   // [256][80]
    const float* emb    = (const float*)d_in[1];   // [10000][100]
    const float* W0     = (const float*)d_in[2];   // [100][2048]
    const float* U0     = (const float*)d_in[3];   // [2048][2048]
    const float* b0     = (const float*)d_in[4];   // [2048]
    const float* W1     = (const float*)d_in[5];   // [2048][2048]
    const float* U1     = (const float*)d_in[6];   // [2048][2048]
    const float* b1     = (const float*)d_in[7];   // [2048]
    const float* Wo     = (const float*)d_in[8];   // [2048]
    const float* bo     = (const float*)d_in[9];   // [1]
    float* out = (float*)d_out;                    // [256]

    // workspace layout (all offsets 256B aligned)
    char* ws = (char*)d_ws;
    unsigned short* Xp  = (unsigned short*)(ws);             //  5,242,880 B
    unsigned short* Wp0 = (unsigned short*)(ws +  5242880);  //  8,912,896 B
    unsigned short* Wp1 = (unsigned short*)(ws + 14155776);  // 16,777,216 B
    float*          Hf  = (float*)         (ws + 30932992);  //  8,388,608 B (4 x 2MB)
    unsigned short* Hp  = (unsigned short*)(ws + 39321600);  //  4,194,304 B (4 x 1MB)

    float* H0a = Hf;
    float* H0b = Hf + BATCH * UNITS;
    float* H1a = Hf + 2 * BATCH * UNITS;
    float* H1b = Hf + 3 * BATCH * UNITS;
    unsigned short* H0ap = Hp;
    unsigned short* H0bp = Hp + 16 * KCH * 512;
    unsigned short* H1ap = Hp + 2 * 16 * KCH * 512;
    unsigned short* H1bp = Hp + 3 * 16 * KCH * 512;

    constexpr int KC0 = 4 + KCH;     // x_t (128 pad) + h0      = 68
    constexpr int KC1 = KCH + KCH;   // h0_new + h1             = 128

    // ---- prep: pack x_t A-fragments (all t), zero state, pack weights ----
    embed_pack_kernel<<<(SEQ * 16 * 4 * 512 + 255) / 256, 256, 0, stream>>>(inputs, emb, Xp);
    zero_hp_kernel<<<(16 * KCH * 256 + 255) / 256, 256, 0, stream>>>(
        (unsigned int*)H0ap, (unsigned int*)H1ap);
    {
        int blk;
        blk = ((UNITS / 16) * 4  * 512 + 255) / 256;
        pack_b_kernel<<<blk, 256, 0, stream>>>(W0, EMBED, 4, 0, KC0, Wp0);
        blk = ((UNITS / 16) * 64 * 512 + 255) / 256;
        pack_b_kernel<<<blk, 256, 0, stream>>>(U0, UNITS, 64, 4,  KC0, Wp0);
        pack_b_kernel<<<blk, 256, 0, stream>>>(W1, UNITS, 64, 0,  KC1, Wp1);
        pack_b_kernel<<<blk, 256, 0, stream>>>(U1, UNITS, 64, 64, KC1, Wp1);
    }

    // ---- recurrence: 2 prepacked WMMA GEMMs per timestep, ping-pong ----
    for (int t = 0; t < SEQ; ++t) {
        unsigned short* h0cp = (t & 1) ? H0bp : H0ap;
        unsigned short* h0np = (t & 1) ? H0ap : H0bp;
        unsigned short* h1cp = (t & 1) ? H1bp : H1ap;
        unsigned short* h1np = (t & 1) ? H1ap : H1bp;
        float* h0n = (t & 1) ? H0a : H0b;
        float* h1n = (t & 1) ? H1a : H1b;
        // layer 0: tanh([x_t | h0] @ [W0;U0] + b0)
        rnn_gemm_tanh<KC0, 4><<<16, 512, 0, stream>>>(
            Xp + (size_t)t * 16 * 4 * 512, h0cp, Wp0, b0, h0n, h0np);
        // layer 1: tanh([h0_new | h1] @ [W1;U1] + b1)
        rnn_gemm_tanh<KC1, KCH><<<16, 512, 0, stream>>>(
            h0np, h1cp, Wp1, b1, h1n, h1np);
    }

    // final H1 fp32 was written by the t=79 layer-1 GEMM into H1a
    final_kernel<<<1, 256, 0, stream>>>(H1a, Wo, bo, out);
}
